// AdaptiveSurBiGCN_5884105195896
// MI455X (gfx1250) — matrix operand order repaired and checked by reading, not compiled, not run on hardware
//
#include <hip/hip_runtime.h>
#include <math.h>

typedef _Float16 f16;
typedef __attribute__((ext_vector_type(16))) _Float16 v16h;
typedef __attribute__((ext_vector_type(8)))  _Float16 v8h;
typedef __attribute__((ext_vector_type(8)))  float    v8f;

#define F 128      // F_IN == H == 128
#define MROWS 80   // rows per block = 5 M-tiles; 50000 = 625 * 80 exactly

// Hardware f32 atomic add (no-return form, tracked by STOREcnt; s_endpgm
// performs an implicit wait-idle). Guaranteed global_atomic_add_f32 — no
// CAS-loop fallback possible.
__device__ __forceinline__ void atomic_add_f32_hw(float* addr, float val) {
  asm volatile("global_atomic_add_f32 %0, %1, off"
               :: "v"(addr), "v"(val) : "memory");
}

// ---------------------------------------------------------------------------
// out[dst[e]] += w[e] * h[src[e]]  — one wave per edge, float4 per lane.
// Feature table (25.6 MB) is L2-resident (192 MB L2); atomics resolve in the
// L2 atomic units. 4 atomics per lane share one address pair via imm offsets.
// ---------------------------------------------------------------------------
__global__ __launch_bounds__(256)
void spmm_edges(const float* __restrict__ h, const int* __restrict__ src,
                const int* __restrict__ dst, const float* __restrict__ w,
                float* __restrict__ out, int E) {
  const int lane = threadIdx.x & 31;
  const int gw   = (int)((blockIdx.x * 256u + threadIdx.x) >> 5);
  const int nw   = (int)((gridDim.x * 256u) >> 5);
  for (int e = gw; e < E; e += nw) {
    const int   s  = __builtin_amdgcn_readfirstlane(src[e]);
    const int   d  = __builtin_amdgcn_readfirstlane(dst[e]);
    const float wt = w[e];
    const float4 v = *(const float4*)(h + (size_t)s * F + lane * 4);
    float* o = out + (size_t)d * F + lane * 4;
    asm volatile(
        "global_atomic_add_f32 %0, %1, off\n\t"
        "global_atomic_add_f32 %0, %2, off offset:4\n\t"
        "global_atomic_add_f32 %0, %3, off offset:8\n\t"
        "global_atomic_add_f32 %0, %4, off offset:12"
        :: "v"(o), "v"(wt * v.x), "v"(wt * v.y), "v"(wt * v.z), "v"(wt * v.w)
        : "memory");
  }
}

// ---------------------------------------------------------------------------
// out[M x 128] = relu(in[M x 128] @ W[128 x 128] + b)
// 256 threads = 8 waves per block; block computes an 80-row x 128-col stripe.
// Wave w owns N-tile [16w, 16w+16) across 5 M-tiles; K loop = 4 steps of 32.
// Per wave: 20 v_wmma_f32_16x16x32_f16, B fragment reused across 5 M-tiles.
// A fragment (16-bit A 16x32): lane<16 holds row=lane, K = kb+{0..7,16..23};
//                              lane>=16 holds row=lane-16, K = kb+{8..15,24..31}.
// B fragment (32x16): lane holds column N = lane&15, K same as A fragment (Ws stored transposed [N][K]).
// C/D (16x16 f32): vgpr g: M = g + 8*(lane>>4), N = lane&15.
// ---------------------------------------------------------------------------
__global__ __launch_bounds__(256)
void gemm_relu_wmma(const float* __restrict__ in, const float* __restrict__ W,
                    const float* __restrict__ b, float* __restrict__ out) {
  __shared__ __align__(32) f16 Ws[F * F];       // 32 KB: whole weight matrix
  __shared__ __align__(32) f16 As[MROWS * F];   // 20 KB: 80 activation rows

  const int tid  = threadIdx.x;
  const int row0 = blockIdx.x * MROWS;

  // cooperative load + f32->f16 convert (coalesced)
  for (int i = tid; i < F * F; i += 256) Ws[i] = (f16)W[(i % F) * F + i / F];
  for (int i = tid; i < MROWS * F; i += 256) As[i] = (f16)in[(size_t)row0 * F + i];
  __syncthreads();

  const int wave = tid >> 5;      // N tile index 0..7
  const int lane = tid & 31;
  const int arow = lane & 15;
  const int koff = (lane < 16) ? 0 : 8;

  union { v16h v; v8h h[2]; } a, bm;
  v8f c[5] = {{}, {}, {}, {}, {}};

#pragma unroll
  for (int kk = 0; kk < 4; ++kk) {
    const int kb = kk * 32;
    // B: column N = wave*16+arow, K = kb+koff+{0..7,16..23} (2x ds_load_b128)
    bm.h[0] = *(const v8h*)&Ws[(wave * 16 + arow) * F + kb + koff];
    bm.h[1] = *(const v8h*)&Ws[(wave * 16 + arow) * F + kb + koff + 16];
#pragma unroll
    for (int mt = 0; mt < 5; ++mt) {
      // A: two 16-byte LDS loads (ds_load_b128)
      const int r = (mt * 16 + arow) * F + kb + koff;
      a.h[0] = *(const v8h*)&As[r];
      a.h[1] = *(const v8h*)&As[r + 16];
      c[mt] = __builtin_amdgcn_wmma_f32_16x16x32_f16(false, a.v, false, bm.v,
                                                     (short)0, c[mt], false, false);
    }
  }

  const int col = wave * 16 + (lane & 15);
  const float bv = b[col];
  const int rsel = (lane >> 4) << 3;
#pragma unroll
  for (int mt = 0; mt < 5; ++mt) {
    const int rbase = row0 + mt * 16 + rsel;
#pragma unroll
    for (int g = 0; g < 8; ++g) {
      float v = c[mt][g] + bv;
      out[(size_t)(rbase + g) * F + col] = v > 0.0f ? v : 0.0f;
    }
  }
}

__global__ void zero_f32(float* __restrict__ p, int n4) {  // n4 = float4 count
  int i  = blockIdx.x * blockDim.x + threadIdx.x;
  int st = gridDim.x * blockDim.x;
  float4 z = {0.f, 0.f, 0.f, 0.f};
  for (; i < n4; i += st) ((float4*)p)[i] = z;
}

// g[c] += sum_r h[r][c]  (per-column partial sums, then atomic combine)
__global__ __launch_bounds__(128)
void colsum(const float* __restrict__ h, float* __restrict__ g, int n) {
  const int c = threadIdx.x;
  float s = 0.f;
  for (int r = blockIdx.x; r < n; r += gridDim.x) s += h[(size_t)r * F + c];
  atomic_add_f32_hw(g + c, s);
}

// mean -> relu(g@Wf1+bf1) -> sigmoid(o@Wf2+bf2)
__global__ __launch_bounds__(128)
void head(const float* __restrict__ g, const float* __restrict__ Wf1,
          const float* __restrict__ bf1, const float* __restrict__ Wf2,
          const float* __restrict__ bf2, float* __restrict__ out, float invN) {
  __shared__ float sg[F];
  __shared__ float so[32];
  const int t = threadIdx.x;
  sg[t] = g[t] * invN;
  __syncthreads();
  if (t < 32) {
    float acc = bf1[t];
#pragma unroll 4
    for (int k = 0; k < F; ++k) acc += sg[k] * Wf1[k * 32 + t];
    so[t] = acc > 0.f ? acc : 0.f;
  }
  __syncthreads();
  if (t == 0) {
    float acc = bf2[0];
#pragma unroll
    for (int k = 0; k < 32; ++k) acc += so[k] * Wf2[k];
    out[0] = 1.0f / (1.0f + expf(-acc));
  }
}

extern "C" void kernel_launch(void* const* d_in, const int* in_sizes, int n_in,
                              void* d_out, int out_size, void* d_ws, size_t ws_size,
                              hipStream_t stream) {
  const float* x    = (const float*)d_in[0];
  const int*   esrc = (const int*)d_in[1];
  const int*   edst = (const int*)d_in[2];
  const float* ew   = (const float*)d_in[3];
  const float* W1   = (const float*)d_in[4];
  const float* b1   = (const float*)d_in[5];
  const float* W2   = (const float*)d_in[6];
  const float* b2   = (const float*)d_in[7];
  const float* W3   = (const float*)d_in[8];
  const float* b3   = (const float*)d_in[9];
  const float* Wf1  = (const float*)d_in[10];
  const float* bf1  = (const float*)d_in[11];
  const float* Wf2  = (const float*)d_in[12];
  const float* bf2  = (const float*)d_in[13];
  float* out = (float*)d_out;

  const int N = in_sizes[0] / F;   // 50000 (divisible by 80)
  const int E = in_sizes[1];       // 1.6M

  float* bufA = (float*)d_ws;
  float* bufB = bufA + (size_t)N * F;
  float* gvec = bufB + (size_t)N * F;

  const int mblocks    = (N + MROWS - 1) / MROWS;  // 625
  const int spmmBlocks = (E + 7) / 8;              // 8 waves (edges) per block
  const int nf4        = (N * F) / 4;

  // layer 1
  gemm_relu_wmma<<<mblocks, 256, 0, stream>>>(x, W1, b1, bufA);
  zero_f32<<<2048, 256, 0, stream>>>(bufB, nf4);
  spmm_edges<<<spmmBlocks, 256, 0, stream>>>(bufA, esrc, edst, ew, bufB, E);
  // layer 2
  gemm_relu_wmma<<<mblocks, 256, 0, stream>>>(bufB, W2, b2, bufA);
  zero_f32<<<2048, 256, 0, stream>>>(bufB, nf4);
  spmm_edges<<<spmmBlocks, 256, 0, stream>>>(bufA, esrc, edst, ew, bufB, E);
  // layer 3
  gemm_relu_wmma<<<mblocks, 256, 0, stream>>>(bufB, W3, b3, bufA);
  zero_f32<<<2048, 256, 0, stream>>>(bufB, nf4);
  spmm_edges<<<spmmBlocks, 256, 0, stream>>>(bufA, esrc, edst, ew, bufB, E);
  // pool + head
  zero_f32<<<1, 32, 0, stream>>>(gvec, F / 4);
  colsum<<<512, 128, 0, stream>>>(bufB, gvec, N);
  head<<<1, 128, 0, stream>>>(gvec, Wf1, bf1, Wf2, bf2, out, 1.0f / (float)N);
}
